// multi_head_attention_72438918414904
// MI455X (gfx1250) — compile-verified, matrix operand order checked by reading, not run on hardware
//
#include <hip/hip_runtime.h>

// ---------------------------------------------------------------------------
// Multi-head attention, MI455X (gfx1250, wave32, WMMA, async global->LDS).
//   0) x -> bf16; Wq/Wk/Wv/Wo -> bf16 transposed [out][in]   (one-time)
//   1) Q,K = bf16 GEMM; V = bf16 GEMM written in V^T layout [b,h,d][s]
//   2) flash attention, 64-key blocks (bf16 WMMA, fp32 accum,
//      rowsum via P@ones WMMA, rowmax via 16-lane butterfly)
//   3) out = Ctx @ Wo + bo (fp32 out)
// Staging: GLOBAL_LOAD_ASYNC_TO_LDS_B128 + s_wait_asynccnt double buffering.
// ---------------------------------------------------------------------------

#define DEV __device__ __forceinline__

typedef __attribute__((ext_vector_type(16))) __bf16 bf16x16;
typedef __attribute__((ext_vector_type(8)))  __bf16 bf16x8;
typedef __attribute__((ext_vector_type(4)))  __bf16 bf16x4;
typedef __attribute__((ext_vector_type(2)))  __bf16 bf16x2;
typedef __attribute__((ext_vector_type(8)))  float  v8f;
typedef __attribute__((ext_vector_type(4)))  int    i32x4;

static constexpr int D     = 1024;
static constexpr int NH    = 16;
static constexpr int HD    = 64;
static constexpr int SEQ   = 2048;
static constexpr int BATCH = 2;
static constexpr int NROW  = BATCH * SEQ; // 4096

// ---- async global->LDS (ASYNCcnt) with sync fallback ----------------------
#if defined(__AMDGCN__) && __has_builtin(__builtin_amdgcn_global_load_async_to_lds_b128)
#define HAS_ASYNC_LDS 1
#else
#define HAS_ASYNC_LDS 0
#endif

#if HAS_ASYNC_LDS
#if __has_builtin(__builtin_amdgcn_s_wait_asynccnt)
#define WAIT_ASYNC(N) __builtin_amdgcn_s_wait_asynccnt(N)
#else
#define WAIT_ASYNC(N) asm volatile("s_wait_asynccnt " #N ::: "memory")
#endif
#else
#define WAIT_ASYNC(N) do {} while (0)
#endif

DEV void async_copy16(const __bf16* gsrc, __bf16* ldst) {
#if HAS_ASYNC_LDS
  __builtin_amdgcn_global_load_async_to_lds_b128(
      (__attribute__((address_space(1))) i32x4*)gsrc,
      (__attribute__((address_space(3))) i32x4*)ldst, 0, 0);
#else
  *(bf16x8*)ldst = *(const bf16x8*)gsrc;
#endif
}

// ---- helpers --------------------------------------------------------------
DEV __bf16 f2bf(float f) {
  unsigned u = __builtin_bit_cast(unsigned, f);
  unsigned r = u + 0x7fffu + ((u >> 16) & 1u);
  unsigned short h = (unsigned short)(r >> 16);
  return __builtin_bit_cast(__bf16, h);
}

// paired f32 -> bf16 (v_cvt_pk_bf16_f32 when available)
#if defined(__AMDGCN__) && __has_builtin(__builtin_amdgcn_cvt_pk_bf16_f32)
DEV bf16x2 pk_bf16(float a, float b) {
  auto r = __builtin_amdgcn_cvt_pk_bf16_f32(a, b);
  return __builtin_bit_cast(bf16x2, r);
}
#else
DEV bf16x2 pk_bf16(float a, float b) {
  bf16x2 r; r[0] = f2bf(a); r[1] = f2bf(b); return r;
}
#endif

DEV v8f zero8() {
  v8f z;
#pragma unroll
  for (int i = 0; i < 8; ++i) z[i] = 0.0f;
  return z;
}

// 16-half WMMA fragment: two contiguous 16B runs at p and p+16 halves
// (CDNA5 16-bit A/B layout: runs at K=kb and K=kb+16, kb = lane<16 ? 0 : 8).
DEV bf16x16 ld_frag(const __bf16* p) {
  bf16x8 lo = *(const bf16x8*)(p);
  bf16x8 hi = *(const bf16x8*)(p + 16);
  bf16x16 r;
#pragma unroll
  for (int i = 0; i < 8; ++i) { r[i] = lo[i]; r[i + 8] = hi[i]; }
  return r;
}

DEV bf16x16 ones_frag() {
  __bf16 one = __builtin_bit_cast(__bf16, (unsigned short)0x3F80);
  bf16x16 r;
#pragma unroll
  for (int i = 0; i < 16; ++i) r[i] = one;
  return r;
}

DEV v8f wmma_bf16(bf16x16 a, bf16x16 b, v8f c) {
  return __builtin_amdgcn_wmma_f32_16x16x32_bf16(false, a, false, b,
                                                 (short)0, c, false, false);
}

// ---------------------------------------------------------------------------
// One-time converts: x fp32 -> bf16; W fp32 [k][n] -> bf16 transposed [n][k].
// ---------------------------------------------------------------------------
__global__ __launch_bounds__(256)
void convert_kernel(const float* __restrict__ in, __bf16* __restrict__ out) {
  size_t i = ((size_t)blockIdx.x * 256 + threadIdx.x) * 8;
  float4 f0 = *(const float4*)(in + i);
  float4 f1 = *(const float4*)(in + i + 4);
  bf16x8 o;
  bf16x2 p0 = pk_bf16(f0.x, f0.y), p1 = pk_bf16(f0.z, f0.w);
  bf16x2 p2 = pk_bf16(f1.x, f1.y), p3 = pk_bf16(f1.z, f1.w);
  o[0] = p0[0]; o[1] = p0[1]; o[2] = p1[0]; o[3] = p1[1];
  o[4] = p2[0]; o[5] = p2[1]; o[6] = p3[0]; o[7] = p3[1];
  *(bf16x8*)(out + i) = o;
}

__global__ __launch_bounds__(256)
void transpose_convert_kernel(const float* __restrict__ in, __bf16* __restrict__ out) {
  __shared__ float tile[32][33];
  const int t  = threadIdx.x;
  const int r  = t >> 3;          // 0..31
  const int c4 = (t & 7) * 4;     // 0..28
  const int tr = blockIdx.y * 32;
  const int tc = blockIdx.x * 32;
  float4 f = *(const float4*)(in + (size_t)(tr + r) * D + tc + c4);
  tile[r][c4 + 0] = f.x; tile[r][c4 + 1] = f.y;
  tile[r][c4 + 2] = f.z; tile[r][c4 + 3] = f.w;
  __syncthreads();
  bf16x2 q0 = pk_bf16(tile[c4 + 0][r], tile[c4 + 1][r]);
  bf16x2 q1 = pk_bf16(tile[c4 + 2][r], tile[c4 + 3][r]);
  bf16x4 o; o[0] = q0[0]; o[1] = q0[1]; o[2] = q1[0]; o[3] = q1[1];
  *(bf16x4*)(out + (size_t)(tc + r) * D + tr + c4) = o;
}

// ---------------------------------------------------------------------------
// GEMM: Out[4096x1024] = A(bf16) @ Wt(bf16,[n][k])^T + bias
// OUT_MODE: 0 = bf16 row-major, 1 = fp32 row-major, 2 = bf16 V^T [b,h,d][s]
// 256 threads (8 waves, tiled 4x2); macro tile 128x128; wave tile 32x64
// (2 A frags x 4 B frags = 8 WMMA, 6 fragment loads); K chunk 32;
// double-buffered async LDS staging (4 async b128 per thread per chunk).
// ---------------------------------------------------------------------------
DEV void stage_tile(const __bf16* gbase, int rowBase, int k0, __bf16* lds, int t) {
#pragma unroll
  for (int p = 0; p < 2; ++p) {
    int idx = p * 256 + t;           // 0..511, 4 threads per 32-wide row
    int r   = idx >> 2;
    int kc  = (idx & 3) * 8;
    async_copy16(gbase + (size_t)(rowBase + r) * D + k0 + kc, &lds[r * 40 + kc]);
  }
}

template<int OUT_MODE>
__global__ __launch_bounds__(256)
void gemm_kernel(const __bf16* __restrict__ A, const __bf16* __restrict__ Wt,
                 const float* __restrict__ bias, void* __restrict__ Out) {
  __shared__ __bf16 As[2][128 * 40];   // [row][k], stride 40 halves (80B)
  __shared__ __bf16 Bs[2][128 * 40];   // [n][k]

  const int t    = threadIdx.x;
  const int wv   = t >> 5;
  const int wr   = wv & 3;             // 4 row groups of 32
  const int wc   = wv >> 2;            // 2 col groups of 64
  const int lane = t & 31;
  const int l16  = lane & 15;
  const int kb   = (lane < 16) ? 0 : 8;
  const int rowBase = blockIdx.y * 128;
  const int colBase = blockIdx.x * 128;

  v8f acc[2][4];
#pragma unroll
  for (int i = 0; i < 2; ++i)
#pragma unroll
    for (int j = 0; j < 4; ++j) acc[i][j] = zero8();

  stage_tile(A,  rowBase, 0, As[0], t);
  stage_tile(Wt, colBase, 0, Bs[0], t);

  constexpr int NCHUNK = D / 32;       // 32
  for (int c = 0; c < NCHUNK; ++c) {
    const int p = c & 1;
    if (c + 1 < NCHUNK) {
      stage_tile(A,  rowBase, (c + 1) * 32, As[1 - p], t);
      stage_tile(Wt, colBase, (c + 1) * 32, Bs[1 - p], t);
      WAIT_ASYNC(4);                   // chunk c landed; chunk c+1 in flight
    } else {
      WAIT_ASYNC(0);
    }
    __syncthreads();

    bf16x16 af0 = ld_frag(&As[p][(wr * 32 +      l16) * 40 + kb]);
    bf16x16 af1 = ld_frag(&As[p][(wr * 32 + 16 + l16) * 40 + kb]);
#pragma unroll
    for (int ct = 0; ct < 4; ++ct) {
      bf16x16 bfrag = ld_frag(&Bs[p][(wc * 64 + ct * 16 + l16) * 40 + kb]);
      acc[0][ct] = wmma_bf16(af0, bfrag, acc[0][ct]);
      acc[1][ct] = wmma_bf16(af1, bfrag, acc[1][ct]);
    }
    __syncthreads();                   // buffer p reusable for chunk c+2
  }

  // epilogue (C layout: lane owns col l16; rows r, +8 for hi lanes)
  const int rhalf = (lane < 16) ? 0 : 8;
#pragma unroll
  for (int i = 0; i < 2; ++i) {
#pragma unroll
    for (int ct = 0; ct < 4; ++ct) {
      int col = colBase + wc * 64 + ct * 16 + l16;
      float bv = bias[col];
      int rowT = rowBase + wr * 32 + i * 16 + rhalf;
#pragma unroll
      for (int r = 0; r < 8; r += 2) {
        float v0 = acc[i][ct][r]     + bv;
        float v1 = acc[i][ct][r + 1] + bv;
        int row0 = rowT + r, row1 = rowT + r + 1;
        if (OUT_MODE == 1) {
          ((float*)Out)[(size_t)row0 * D + col] = v0;
          ((float*)Out)[(size_t)row1 * D + col] = v1;
        } else if (OUT_MODE == 0) {
          bf16x2 pr = pk_bf16(v0, v1);
          ((__bf16*)Out)[(size_t)row0 * D + col] = pr[0];
          ((__bf16*)Out)[(size_t)row1 * D + col] = pr[1];
        } else { // V^T layout: [b, h, d][s]
          int hh = col >> 6, dd = col & 63;
          int b0 = row0 >> 11, s0 = row0 & 2047;
          int b1 = row1 >> 11, s1 = row1 & 2047;
          bf16x2 pr = pk_bf16(v0, v1);
          ((__bf16*)Out)[((size_t)(b0 * NH + hh) * HD + dd) * SEQ + s0] = pr[0];
          ((__bf16*)Out)[((size_t)(b1 * NH + hh) * HD + dd) * SEQ + s1] = pr[1];
        }
      }
    }
  }
}

// ---------------------------------------------------------------------------
// Flash attention. Grid (S/128, B*H), block 256 (8 waves), 64-key blocks.
// K staged row-major [n][d]; V staged from pre-transposed V^T as [d][n];
// both async, double-buffered. Row sums via WMMA against all-ones fragment.
// ---------------------------------------------------------------------------
DEV void stage_kv(const __bf16* K, const __bf16* VtBase, size_t rowOff,
                  int colH, int n0, __bf16* ksh, __bf16* vsh, int t) {
#pragma unroll
  for (int p = 0; p < 2; ++p) {
    int idx = p * 256 + t;             // 0..511: 64 rows x 8 groups of 8
    int rr  = idx >> 3;
    int cc  = (idx & 7) * 8;
    async_copy16(&K[(rowOff + n0 + rr) * D + colH + cc], &ksh[rr * 72 + cc]);
    async_copy16(&VtBase[(size_t)rr * SEQ + n0 + cc],    &vsh[rr * 72 + cc]);
  }
}

__global__ __launch_bounds__(256)
void flash_kernel(const __bf16* __restrict__ Q, const __bf16* __restrict__ K,
                  const __bf16* __restrict__ Vt, __bf16* __restrict__ Ctx) {
  __shared__ __bf16 Ksh[2][64 * 72];     // [n][d], stride 72 halves (144B)
  __shared__ __bf16 Vsh[2][64 * 72];     // [d][n]
  __shared__ __bf16 Psh[8 * 16 * 72];    // per-wave P scratch [row][n 0..63]

  const int t    = threadIdx.x;
  const int wv   = t >> 5;
  const int lane = t & 31;
  const int l16  = lane & 15;
  const int kb   = (lane < 16) ? 0 : 8;

  const int q0   = blockIdx.x * 128 + wv * 16;
  const int bh   = blockIdx.y;
  const int b    = bh >> 4;
  const int h    = bh & 15;
  const size_t rowOff = (size_t)b * SEQ;
  const int colH = h * HD;
  const __bf16* VtBase = Vt + (size_t)bh * HD * SEQ;

  // Q fragments live in registers for the whole key sweep
  bf16x16 qf[2];
#pragma unroll
  for (int dc = 0; dc < 2; ++dc)
    qf[dc] = ld_frag(&Q[(rowOff + q0 + l16) * D + colH + dc * 32 + kb]);

  const bf16x16 onef = ones_frag();

  v8f oacc[4];
#pragma unroll
  for (int i = 0; i < 4; ++i) oacc[i] = zero8();
  float m[8], l[8];
#pragma unroll
  for (int r = 0; r < 8; ++r) { m[r] = -INFINITY; l[r] = 0.0f; }

  __bf16* myP = &Psh[wv * 16 * 72];

  stage_kv(K, VtBase, rowOff, colH, 0, Ksh[0], Vsh[0], t);

  constexpr int NCHUNK = SEQ / 64;   // 32
  for (int c = 0; c < NCHUNK; ++c) {
    const int p = c & 1;
    if (c + 1 < NCHUNK) {
      stage_kv(K, VtBase, rowOff, colH, (c + 1) * 64, Ksh[1 - p], Vsh[1 - p], t);
      WAIT_ASYNC(4);
    } else {
      WAIT_ASYNC(0);
    }
    __syncthreads();

    // scores for 64 keys: four 16x16 frags
    v8f s[4];
#pragma unroll
    for (int j = 0; j < 4; ++j) s[j] = zero8();
#pragma unroll
    for (int dc = 0; dc < 2; ++dc) {
#pragma unroll
      for (int j = 0; j < 4; ++j) {
        bf16x16 kf = ld_frag(&Ksh[p][(j * 16 + l16) * 72 + dc * 32 + kb]);
        s[j] = wmma_bf16(qf[dc], kf, s[j]);
      }
    }

    // online softmax: row max over 64 keys (3 reg maxes + one butterfly)
    float mnew[8], alpha[8];
#pragma unroll
    for (int r = 0; r < 8; ++r) {
      float a0 = s[0][r] * 0.125f; s[0][r] = a0;   // 1/sqrt(64)
      float a1 = s[1][r] * 0.125f; s[1][r] = a1;
      float a2 = s[2][r] * 0.125f; s[2][r] = a2;
      float a3 = s[3][r] * 0.125f; s[3][r] = a3;
      float mx = fmaxf(fmaxf(a0, a1), fmaxf(a2, a3));
#pragma unroll
      for (int off = 1; off < 16; off <<= 1)
        mx = fmaxf(mx, __shfl_xor(mx, off, 16));
      mnew[r]  = fmaxf(m[r], mx);
      alpha[r] = __expf(m[r] - mnew[r]);
      m[r]     = mnew[r];
    }
    // exponentiate and store P tile (C-frag -> A-frag relayout via LDS)
#pragma unroll
    for (int r = 0; r < 8; ++r) {
      float p0 = __expf(s[0][r] - mnew[r]);
      float p1 = __expf(s[1][r] - mnew[r]);
      float p2 = __expf(s[2][r] - mnew[r]);
      float p3 = __expf(s[3][r] - mnew[r]);
      int row = r + ((lane < 16) ? 0 : 8);
      bf16x2 q01 = pk_bf16(p0, p1);
      bf16x2 q23 = pk_bf16(p2, p3);
      myP[row * 72 +      l16] = q01[0];
      myP[row * 72 + 16 + l16] = q01[1];
      myP[row * 72 + 32 + l16] = q23[0];
      myP[row * 72 + 48 + l16] = q23[1];
    }

    // reload P as two A-fragments (same-wave LDS RAW; hw dscnt wait)
    bf16x16 pf0 = ld_frag(&myP[l16 * 72 +      kb]);
    bf16x16 pf1 = ld_frag(&myP[l16 * 72 + 32 + kb]);

    // row sums via WMMA against ones: D[r][*] = sum_k P[r][k]
    v8f sacc = wmma_bf16(pf0, onef, zero8());
    sacc = wmma_bf16(pf1, onef, sacc);
#pragma unroll
    for (int r = 0; r < 8; ++r) l[r] = l[r] * alpha[r] + sacc[r];

#pragma unroll
    for (int dt = 0; dt < 4; ++dt) {
#pragma unroll
      for (int r = 0; r < 8; ++r) oacc[dt][r] *= alpha[r];
    }

    // O += P @ V (two 32-key halves per head-dim tile)
#pragma unroll
    for (int dt = 0; dt < 4; ++dt) {
      bf16x16 vf0 = ld_frag(&Vsh[p][(dt * 16 + l16) * 72 +      kb]);
      bf16x16 vf1 = ld_frag(&Vsh[p][(dt * 16 + l16) * 72 + 32 + kb]);
      oacc[dt] = wmma_bf16(pf0, vf0, oacc[dt]);
      oacc[dt] = wmma_bf16(pf1, vf1, oacc[dt]);
    }
    __syncthreads();
  }

  // epilogue: O / l -> bf16 context [token][h*64+d]
  const int rhalf = (lane < 16) ? 0 : 8;
#pragma unroll
  for (int r = 0; r < 8; r += 2) {
    float i0 = 1.0f / l[r];
    float i1 = 1.0f / l[r + 1];
    int row0 = q0 + r + rhalf;
    int row1 = row0 + 1;
#pragma unroll
    for (int dt = 0; dt < 4; ++dt) {
      int col = colH + dt * 16 + l16;
      bf16x2 pr = pk_bf16(oacc[dt][r] * i0, oacc[dt][r + 1] * i1);
      Ctx[(rowOff + row0) * D + col] = pr[0];
      Ctx[(rowOff + row1) * D + col] = pr[1];
    }
  }
}

// ---------------------------------------------------------------------------
extern "C" void kernel_launch(void* const* d_in, const int* in_sizes, int n_in,
                              void* d_out, int out_size, void* d_ws, size_t ws_size,
                              hipStream_t stream) {
  const float* x  = (const float*)d_in[0];
  const float* Wq = (const float*)d_in[1];
  const float* bq = (const float*)d_in[2];
  const float* Wk = (const float*)d_in[3];
  const float* bk = (const float*)d_in[4];
  const float* Wv = (const float*)d_in[5];
  const float* bv = (const float*)d_in[6];
  const float* Wo = (const float*)d_in[7];
  const float* bo = (const float*)d_in[8];
  float* out = (float*)d_out;

  // workspace (bf16): xb,Qb,Kb,Vt,Ctx = 8 MiB each; 4 transposed weights 2 MiB
  char* ws = (char*)d_ws;
  const size_t MB = 1024ull * 1024ull;
  __bf16* xb  = (__bf16*)(ws);
  __bf16* Qb  = (__bf16*)(ws + 8  * MB);
  __bf16* Kb  = (__bf16*)(ws + 16 * MB);
  __bf16* Vt  = (__bf16*)(ws + 24 * MB);
  __bf16* Cb  = (__bf16*)(ws + 32 * MB);
  __bf16* Wqt = (__bf16*)(ws + 40 * MB);
  __bf16* Wkt = (__bf16*)(ws + 42 * MB);
  __bf16* Wvt = (__bf16*)(ws + 44 * MB);
  __bf16* Wot = (__bf16*)(ws + 46 * MB);

  convert_kernel<<<dim3((unsigned)((size_t)NROW * D / (256 * 8))), 256, 0, stream>>>(x, xb);
  dim3 tg(D / 32, D / 32);
  transpose_convert_kernel<<<tg, 256, 0, stream>>>(Wq, Wqt);
  transpose_convert_kernel<<<tg, 256, 0, stream>>>(Wk, Wkt);
  transpose_convert_kernel<<<tg, 256, 0, stream>>>(Wv, Wvt);
  transpose_convert_kernel<<<tg, 256, 0, stream>>>(Wo, Wot);

  dim3 gg(D / 128, NROW / 128);     // (8, 32)
  gemm_kernel<0><<<gg, 256, 0, stream>>>(xb, Wqt, bq, Qb);
  gemm_kernel<0><<<gg, 256, 0, stream>>>(xb, Wkt, bk, Kb);
  gemm_kernel<2><<<gg, 256, 0, stream>>>(xb, Wvt, bv, Vt);

  dim3 fg(SEQ / 128, BATCH * NH);   // (16, 32)
  flash_kernel<<<fg, 256, 0, stream>>>(Qb, Kb, Vt, Cb);

  gemm_kernel<1><<<gg, 256, 0, stream>>>(Cb, Wot, bo, out);
}
